// GAT_17489106829855
// MI455X (gfx1250) — compile-verified, matrix operand order checked by reading, not run on hardware
//
#include <hip/hip_runtime.h>
#include <hip/hip_bf16.h>
#include <math.h>

#define N_NODES  50000
#define N_EDGES  800000
#define ETOT     (N_EDGES + N_NODES)   // 850000 (self-loops appended)
#define IN_CH    128
#define HEADS    16
#define HID      16
#define C1       (HEADS * HID)         // 256
#define OUT_CH   2
#define N_GRAPHS 512
#define NEG_SLOPE 0.2f

typedef __attribute__((ext_vector_type(16))) __bf16 v16bf;
typedef __attribute__((ext_vector_type(8)))  float  v8f;

// ---------------- helpers ----------------

__device__ inline void split_bf16(float f, __bf16& hi, __bf16& lo) {
  __bf16 h = (__bf16)f;
  hi = h;
  lo = (__bf16)(f - (float)h);
}

__device__ inline float lrelu(float a) { return a > 0.0f ? a : NEG_SLOPE * a; }

// f32 atomic max via sign-aware integer atomics (init must be -inf)
__device__ inline void atomicMaxF(float* addr, float v) {
  if (v >= 0.0f) atomicMax((int*)addr, __float_as_int(v));
  else           atomicMin((unsigned int*)addr, (unsigned int)__float_as_int(v));
}

__device__ inline void edge_sd(int e, const int* __restrict__ ei, int& s, int& d) {
  if (e < N_EDGES) { s = ei[e]; d = ei[N_EDGES + e]; }
  else             { s = e - N_EDGES; d = s; }          // self loop
}

// ---------------- kernels ----------------

__global__ void k_fill(float* __restrict__ p, float v, int n) {
  int i = blockIdx.x * blockDim.x + threadIdx.x;
  if (i < n) p[i] = v;
}

// GEMM1: xw = x @ W1   [50000 x 128] * [128 x 256]
// Split-precision bf16 WMMA: acc += Ah*Bh + Al*Bh + Ah*Bl (f32 accumulate).
// Block = 128 threads = 4 waves; block covers 16 rows, wave covers 64 cols (4 C-tiles).
__global__ __launch_bounds__(128) void k_gemm1(const float* __restrict__ x,
                                               const float* __restrict__ W,
                                               float* __restrict__ xw) {
  const int lane = threadIdx.x & 31;
  const int wave = threadIdx.x >> 5;     // 0..3
  const int half = lane >> 4;            // selects K half per ISA layout
  const int l15  = lane & 15;
  const int m0   = blockIdx.x * 16;
  const int nb   = wave * 64;

  v8f acc[4];
#pragma unroll
  for (int t = 0; t < 4; ++t)
#pragma unroll
    for (int r = 0; r < 8; ++r) acc[t][r] = 0.0f;

  const float* arow = x + (size_t)(m0 + l15) * IN_CH;

#pragma unroll
  for (int ks = 0; ks < 4; ++ks) {
    const int k0 = ks * 32;
    // A fragment 16x32 bf16: per-lane K runs {half*8+0..7, 16+half*8+0..7}
    float af[16];
    const float* ap = arow + k0 + half * 8;
#pragma unroll
    for (int i = 0; i < 8; ++i) af[i]     = ap[i];
#pragma unroll
    for (int i = 0; i < 8; ++i) af[8 + i] = ap[16 + i];
    v16bf ah, al;
#pragma unroll
    for (int i = 0; i < 16; ++i) { __bf16 hi, lo; split_bf16(af[i], hi, lo); ah[i] = hi; al[i] = lo; }

#pragma unroll
    for (int t = 0; t < 4; ++t) {
      const int n0 = nb + t * 16;
      // B fragment 32x16 bf16: lane = column, same per-lane K mapping as A
      const float* wp = W + (size_t)(k0 + half * 8) * C1 + (n0 + l15);
      float bfv[16];
#pragma unroll
      for (int i = 0; i < 8; ++i) bfv[i]     = wp[(size_t)i * C1];
#pragma unroll
      for (int i = 0; i < 8; ++i) bfv[8 + i] = wp[(size_t)(16 + i) * C1];
      v16bf bh, bl;
#pragma unroll
      for (int i = 0; i < 16; ++i) { __bf16 hi, lo; split_bf16(bfv[i], hi, lo); bh[i] = hi; bl[i] = lo; }

      acc[t] = __builtin_amdgcn_wmma_f32_16x16x32_bf16(false, ah, false, bh, (short)0, acc[t], false, false);
      acc[t] = __builtin_amdgcn_wmma_f32_16x16x32_bf16(false, al, false, bh, (short)0, acc[t], false, false);
      acc[t] = __builtin_amdgcn_wmma_f32_16x16x32_bf16(false, ah, false, bl, (short)0, acc[t], false, false);
    }
  }

  // C/D: VGPR r, lane -> row m0 + r + 8*half, col n0 + l15
#pragma unroll
  for (int t = 0; t < 4; ++t) {
    const int n0 = nb + t * 16;
#pragma unroll
    for (int r = 0; r < 8; ++r)
      xw[(size_t)(m0 + r + 8 * half) * C1 + n0 + l15] = acc[t][r];
  }
}

// Per-(node,head) attention coefficients for conv1
__global__ __launch_bounds__(256) void k_att1(const float* __restrict__ xw,
                                              const float* __restrict__ as1,
                                              const float* __restrict__ ad1,
                                              float* __restrict__ es1,
                                              float* __restrict__ ed1) {
  int idx = blockIdx.x * 256 + threadIdx.x;
  if (idx >= N_NODES * HEADS) return;
  int n = idx >> 4, h = idx & 15;
  const float* row = xw + (size_t)n * C1 + h * HID;
  float a = 0.0f, b = 0.0f;
#pragma unroll
  for (int c = 0; c < HID; ++c) {
    float v = row[c];
    a += v * as1[h * HID + c];
    b += v * ad1[h * HID + c];
  }
  es1[idx] = a;
  ed1[idx] = b;
}

// Edge pass 1: segment max of leaky_relu(e_src[src]+e_dst[dst]) per (dst,head)
__global__ __launch_bounds__(256) void k_edge_max1(const int* __restrict__ ei,
                                                   const float* __restrict__ es,
                                                   const float* __restrict__ ed,
                                                   float* __restrict__ amax) {
  int idx = blockIdx.x * 256 + threadIdx.x;
  if (idx >= ETOT * HEADS) return;
  int e = idx >> 4, h = idx & 15;
  int s, d; edge_sd(e, ei, s, d);
  float a = lrelu(es[s * HEADS + h] + ed[d * HEADS + h]);
  atomicMaxF(&amax[d * HEADS + h], a);
}

// Edge pass 2: denom = segment sum of exp(a - amax[dst])
__global__ __launch_bounds__(256) void k_edge_sum1(const int* __restrict__ ei,
                                                   const float* __restrict__ es,
                                                   const float* __restrict__ ed,
                                                   const float* __restrict__ amax,
                                                   float* __restrict__ den) {
  int idx = blockIdx.x * 256 + threadIdx.x;
  if (idx >= ETOT * HEADS) return;
  int e = idx >> 4, h = idx & 15;
  int s, d; edge_sd(e, ei, s, d);
  float a = lrelu(es[s * HEADS + h] + ed[d * HEADS + h]);
  atomicAdd(&den[d * HEADS + h], __expf(a - amax[d * HEADS + h]));
}

// Edge pass 3: agg[dst] += alpha * xw[src]  (alpha recomputed; 256 ch per edge)
// grid = ETOT blocks x 256 thr; thread r covers head r>>4, channel r&15 (coalesced).
__global__ __launch_bounds__(256) void k_edge_agg1(const int* __restrict__ ei,
                                                   const float* __restrict__ es,
                                                   const float* __restrict__ ed,
                                                   const float* __restrict__ amax,
                                                   const float* __restrict__ den,
                                                   const float* __restrict__ xw,
                                                   float* __restrict__ agg) {
  int e = blockIdx.x;
  int r = threadIdx.x;        // = h*16 + c
  int h = r >> 4;
  int s, d; edge_sd(e, ei, s, d);
  float a = lrelu(es[s * HEADS + h] + ed[d * HEADS + h]);
  float w = __expf(a - amax[d * HEADS + h]) / den[d * HEADS + h];
  atomicAdd(&agg[(size_t)d * C1 + r], w * xw[(size_t)s * C1 + r]);
}

// Fused: h = elu(agg + b1); xw2 = h @ W2; es2/ed2 attention coeffs (1 head, 2 ch).
// One wave per node, shuffle reduction.
__global__ __launch_bounds__(256) void k_conv2lin(const float* __restrict__ agg,
                                                  const float* __restrict__ b1,
                                                  const float* __restrict__ W2,
                                                  const float* __restrict__ as2,
                                                  const float* __restrict__ ad2,
                                                  float* __restrict__ xw2,
                                                  float* __restrict__ es2,
                                                  float* __restrict__ ed2) {
  int wid  = blockIdx.x * (blockDim.x >> 5) + (threadIdx.x >> 5);
  if (wid >= N_NODES) return;
  int lane = threadIdx.x & 31;
  const float* row = agg + (size_t)wid * C1;
  float o0 = 0.0f, o1 = 0.0f;
#pragma unroll
  for (int j = lane; j < C1; j += 32) {
    float v = row[j] + b1[j];
    v = v > 0.0f ? v : (__expf(v) - 1.0f);   // ELU
    o0 += v * W2[j * 2 + 0];
    o1 += v * W2[j * 2 + 1];
  }
#pragma unroll
  for (int m = 16; m; m >>= 1) {
    o0 += __shfl_xor(o0, m, 32);
    o1 += __shfl_xor(o1, m, 32);
  }
  if (lane == 0) {
    xw2[wid * 2 + 0] = o0;
    xw2[wid * 2 + 1] = o1;
    es2[wid] = o0 * as2[0] + o1 * as2[1];
    ed2[wid] = o0 * ad2[0] + o1 * ad2[1];
  }
}

__global__ __launch_bounds__(256) void k_edge_max2(const int* __restrict__ ei,
                                                   const float* __restrict__ es,
                                                   const float* __restrict__ ed,
                                                   float* __restrict__ amax) {
  int e = blockIdx.x * 256 + threadIdx.x;
  if (e >= ETOT) return;
  int s, d; edge_sd(e, ei, s, d);
  atomicMaxF(&amax[d], lrelu(es[s] + ed[d]));
}

__global__ __launch_bounds__(256) void k_edge_sum2(const int* __restrict__ ei,
                                                   const float* __restrict__ es,
                                                   const float* __restrict__ ed,
                                                   const float* __restrict__ amax,
                                                   float* __restrict__ den) {
  int e = blockIdx.x * 256 + threadIdx.x;
  if (e >= ETOT) return;
  int s, d; edge_sd(e, ei, s, d);
  float a = lrelu(es[s] + ed[d]);
  atomicAdd(&den[d], __expf(a - amax[d]));
}

__global__ __launch_bounds__(256) void k_edge_agg2(const int* __restrict__ ei,
                                                   const float* __restrict__ es,
                                                   const float* __restrict__ ed,
                                                   const float* __restrict__ amax,
                                                   const float* __restrict__ den,
                                                   const float* __restrict__ xw2,
                                                   float* __restrict__ out2) {
  int idx = blockIdx.x * 256 + threadIdx.x;
  if (idx >= ETOT * OUT_CH) return;
  int e = idx >> 1, oc = idx & 1;
  int s, d; edge_sd(e, ei, s, d);
  float a = lrelu(es[s] + ed[d]);
  float w = __expf(a - amax[d]) / den[d];
  atomicAdd(&out2[d * OUT_CH + oc], w * xw2[s * OUT_CH + oc]);
}

// global mean pool: sums + counts
__global__ __launch_bounds__(256) void k_pool(const float* __restrict__ out2,
                                              const float* __restrict__ b2,
                                              const int* __restrict__ batch,
                                              float* __restrict__ pooled,
                                              float* __restrict__ cnts) {
  int n = blockIdx.x * 256 + threadIdx.x;
  if (n >= N_NODES) return;
  int g = batch[n];
  atomicAdd(&pooled[g * OUT_CH + 0], out2[n * OUT_CH + 0] + b2[0]);
  atomicAdd(&pooled[g * OUT_CH + 1], out2[n * OUT_CH + 1] + b2[1]);
  atomicAdd(&cnts[g], 1.0f);
}

__global__ __launch_bounds__(256) void k_final(const float* __restrict__ pooled,
                                               const float* __restrict__ cnts,
                                               float* __restrict__ out) {
  int i = blockIdx.x * 256 + threadIdx.x;
  if (i >= N_GRAPHS * OUT_CH) return;
  out[i] = pooled[i] / fmaxf(cnts[i >> 1], 1.0f);
}

// ---------------- launcher ----------------

extern "C" void kernel_launch(void* const* d_in, const int* in_sizes, int n_in,
                              void* d_out, int out_size, void* d_ws, size_t ws_size,
                              hipStream_t stream) {
  const float* x     = (const float*)d_in[0];
  const int*   ei    = (const int*)  d_in[1];
  const int*   batch = (const int*)  d_in[2];
  const float* W1    = (const float*)d_in[3];
  const float* as1   = (const float*)d_in[4];
  const float* ad1   = (const float*)d_in[5];
  const float* b1    = (const float*)d_in[6];
  const float* W2    = (const float*)d_in[7];
  const float* as2   = (const float*)d_in[8];
  const float* ad2   = (const float*)d_in[9];
  const float* b2    = (const float*)d_in[10];
  float* out = (float*)d_out;

  float* ws = (float*)d_ws;
  size_t o = 0;
  float* xw1    = ws + o; o += (size_t)N_NODES * C1;       // 12.8M
  float* es1    = ws + o; o += (size_t)N_NODES * HEADS;
  float* ed1    = ws + o; o += (size_t)N_NODES * HEADS;
  float* amax1  = ws + o; o += (size_t)N_NODES * HEADS;
  float* den1   = ws + o; o += (size_t)N_NODES * HEADS;
  float* agg1   = ws + o; o += (size_t)N_NODES * C1;       // 12.8M
  float* xw2    = ws + o; o += (size_t)N_NODES * OUT_CH;
  float* es2    = ws + o; o += (size_t)N_NODES;
  float* ed2    = ws + o; o += (size_t)N_NODES;
  float* amax2  = ws + o; o += (size_t)N_NODES;
  float* den2   = ws + o; o += (size_t)N_NODES;
  float* out2   = ws + o; o += (size_t)N_NODES * OUT_CH;
  float* pooled = ws + o; o += (size_t)N_GRAPHS * OUT_CH;
  float* cnts   = ws + o; o += (size_t)N_GRAPHS;

  auto fill = [&](float* p, float v, size_t n) {
    k_fill<<<(unsigned)((n + 255) / 256), 256, 0, stream>>>(p, v, (int)n);
  };

  // per-call re-init (graph-replay safe)
  fill(amax1, -INFINITY, (size_t)N_NODES * HEADS);
  fill(den1,  0.0f,      (size_t)N_NODES * HEADS);
  fill(agg1,  0.0f,      (size_t)N_NODES * C1);
  fill(amax2, -INFINITY, (size_t)N_NODES);
  fill(den2,  0.0f,      (size_t)N_NODES);
  fill(out2,  0.0f,      (size_t)N_NODES * OUT_CH);
  fill(pooled,0.0f,      (size_t)N_GRAPHS * OUT_CH);
  fill(cnts,  0.0f,      (size_t)N_GRAPHS);

  // conv1 linear (WMMA) + attention coefficients
  k_gemm1<<<N_NODES / 16, 128, 0, stream>>>(x, W1, xw1);
  k_att1<<<(N_NODES * HEADS + 255) / 256, 256, 0, stream>>>(xw1, as1, ad1, es1, ed1);

  // conv1 edge softmax + aggregation
  k_edge_max1<<<(ETOT * HEADS + 255) / 256, 256, 0, stream>>>(ei, es1, ed1, amax1);
  k_edge_sum1<<<(ETOT * HEADS + 255) / 256, 256, 0, stream>>>(ei, es1, ed1, amax1, den1);
  k_edge_agg1<<<ETOT, 256, 0, stream>>>(ei, es1, ed1, amax1, den1, xw1, agg1);

  // elu+bias, conv2 linear, attention coefficients (fused)
  k_conv2lin<<<(N_NODES + 7) / 8, 256, 0, stream>>>(agg1, b1, W2, as2, ad2, xw2, es2, ed2);

  // conv2 edge softmax + aggregation
  k_edge_max2<<<(ETOT + 255) / 256, 256, 0, stream>>>(ei, es2, ed2, amax2);
  k_edge_sum2<<<(ETOT + 255) / 256, 256, 0, stream>>>(ei, es2, ed2, amax2, den2);
  k_edge_agg2<<<(ETOT * OUT_CH + 255) / 256, 256, 0, stream>>>(ei, es2, ed2, amax2, den2, xw2, out2);

  // global mean pool
  k_pool<<<(N_NODES + 255) / 256, 256, 0, stream>>>(out2, b2, batch, pooled, cnts);
  k_final<<<(N_GRAPHS * OUT_CH + 255) / 256, 256, 0, stream>>>(pooled, cnts, out);
}
